// GAT_49039936586472
// MI455X (gfx1250) — compile-verified, matrix operand order checked by reading, not run on hardware
//
#include <hip/hip_runtime.h>

#define GAT_HEADS 4

typedef __attribute__((ext_vector_type(16))) _Float16 v16h;
typedef __attribute__((ext_vector_type(8)))  float    v8f;

__device__ __forceinline__ float leaky02(float z) { return z > 0.f ? z : 0.2f * z; }

// Order-preserving float -> uint map so we can do float segment-max with atomicMax(u32).
__device__ __forceinline__ unsigned f2ord(float f) {
  unsigned u = __float_as_uint(f);
  return (u >> 31) ? ~u : (u | 0x80000000u);
}
__device__ __forceinline__ float ord2f(unsigned o) {
  return (o >> 31) ? __uint_as_float(o & 0x7fffffffu) : __uint_as_float(~o);
}

// ---------------- WMMA GEMM: f32 data -> f16 fragments, 16x16x32, f32 accum ----------
// 16-bit A-matrix 16x32 layout (ISA 7.12.2): lanes 0-15 -> M=lane, K={0..7,16..23};
// lanes 16-31 -> M=lane-16, K={8..15,24..31}; VGPR j<4: K=8*hi+2j+t, j>=4: K=16+8*hi+2(j-4)+t.
// K is even in all uses, so each (2j,2j+1) pair is a contiguous, 8B-aligned float2 in A.
template<int K>
__device__ __forceinline__ v16h load_a_frag(const float* __restrict__ A, int row0, int M) {
  static_assert((K & 1) == 0, "K must be even for paired loads");
  const int lane = threadIdx.x & 31;
  const int m = lane & 15, hi = lane >> 4;
  const int r = row0 + m;
  const bool rok = (r < M);
  v16h a;
#pragma unroll
  for (int j = 0; j < 8; ++j) {
    const int kb = (j < 4) ? (hi * 8 + 2 * j) : (16 + hi * 8 + 2 * (j - 4));
    const bool ok = rok && (kb < K);           // kb even, K even => pair wholly in/out
    const size_t idx = ok ? ((size_t)r * K + kb) : 0;  // clamp: load always legal, select after
    const float2 v = *(const float2*)(A + idx);
    a[2 * j + 0] = (_Float16)(ok ? v.x : 0.f);
    a[2 * j + 1] = (_Float16)(ok ? v.y : 0.f);
  }
  return a;
}

// B fragment from LDS tile laid out [n][k] (k fastest, 32 ks per column, f16, zero-padded):
// 8 unpredicated ds_load_b32 per lane, no bounds logic (padding is zero-filled).
__device__ __forceinline__ v16h load_b_frag_lds(const _Float16* __restrict__ bsh, int col0) {
  const int lane = threadIdx.x & 31;
  const int n = lane & 15, hi = lane >> 4;
  const _Float16* bp = bsh + (col0 + n) * 32;
  union { v16h h; unsigned u[8]; } bb;
#pragma unroll
  for (int j = 0; j < 8; ++j) {
    const int kb = (j < 4) ? (hi * 8 + 2 * j) : (16 + hi * 8 + 2 * (j - 4));
    bb.u[j] = *(const unsigned*)(bp + kb);     // kb even -> 4B aligned packed pair
  }
  return bb.h;
}

// f32 C/D 16x16 layout: VGPR r -> M = r + 8*(lane>=16), N = lane&15.
// D and the dump sink both live in d_ws, so OOB lanes select a 32-bit ws-relative
// element index (single v_cndmask_b32) instead of a 64-bit pointer -> branch-free,
// no EXEC writes, unpredicated global_store_b32.
template<int NCOLS>
__device__ __forceinline__ void store_d(float* __restrict__ wsbase, unsigned doff,
                                        const v8f& d, int row0, int M, int col0,
                                        unsigned dumpoff) {
  const int lane = threadIdx.x & 31;
  const int n = lane & 15, hi = lane >> 4;
  const int c = col0 + n;
  const bool cok = (c < NCOLS);                // folds to true for full column tiles
  const unsigned bad = dumpoff + (unsigned)lane;
#pragma unroll
  for (int r = 0; r < 8; ++r) {
    const int m = row0 + r + 8 * hi;
    const bool ok = (m < M) && cok;
    const unsigned idx = ok ? (doff + (unsigned)m * NCOLS + (unsigned)c) : bad;
    wsbase[idx] = d[r];
  }
}

// D[M,NCOLS] = A[M,K] @ W[K,NCOLS]; one wave per 16-row tile (wave-uniform guard -> EXEC all 1s).
// D is given as a float-element offset (doff) into the workspace base.
template<int K, int NCOLS>
__global__ void gemm_wmma(const float* __restrict__ A, const float* __restrict__ W,
                          float* __restrict__ wsbase, unsigned doff, int M,
                          unsigned dumpoff) {
  constexpr int NPAD = (NCOLS + 15) & ~15;
  __shared__ __align__(16) _Float16 bsh[NPAD * 32];
  // Cooperative stage of weights into LDS: f16, K zero-padded to 32, N zero-padded to NPAD.
  for (int idx = threadIdx.x; idx < NPAD * 32; idx += blockDim.x) {
    const int k = idx & 31, n = idx >> 5;
    const float v = (k < K && n < NCOLS) ? W[k * NCOLS + n] : 0.f;
    bsh[idx] = (_Float16)v;
  }
  __syncthreads();

  const int wave = (int)((blockIdx.x * blockDim.x + threadIdx.x) >> 5);
  const int row0 = wave * 16;
  if (row0 >= M) return;                        // wave-uniform
  const v16h a = load_a_frag<K>(A, row0, M);
#pragma unroll
  for (int col0 = 0; col0 < NCOLS; col0 += 16) {
    const v16h b = load_b_frag_lds(bsh, col0);
    v8f c = {};
    c = __builtin_amdgcn_wmma_f32_16x16x32_f16(false, a, false, b, (short)0, c, false, false);
    store_d<NCOLS>(wsbase, doff, c, row0, M, col0, dumpoff);
  }
}

// ---------------- scalar / edge kernels ----------------
__global__ void fill_u32(unsigned* __restrict__ p, unsigned v, long long n) {
  long long i = (long long)blockIdx.x * blockDim.x + threadIdx.x;
  if (i < n) p[i] = v;
}

// per-node attention coefficients: asrc[n,h] = sum_c h[n,h,c]*wsrc[h,c] (same for adst)
__global__ void att_coef(const float* __restrict__ h, const float* __restrict__ wsrc,
                         const float* __restrict__ wdst, float* __restrict__ asrc,
                         float* __restrict__ adst, int n, int C) {
  int i = blockIdx.x * blockDim.x + threadIdx.x;
  if (i >= n) return;
  const float* hp = h + (size_t)i * GAT_HEADS * C;
#pragma unroll
  for (int hh = 0; hh < GAT_HEADS; ++hh) {
    float s = 0.f, d = 0.f;
    for (int c = 0; c < C; ++c) {
      float v = hp[hh * C + c];
      s += v * wsrc[hh * C + c];
      d += v * wdst[hh * C + c];
    }
    asrc[(size_t)i * GAT_HEADS + hh] = s;
    adst[(size_t)i * GAT_HEADS + hh] = d;
  }
}

// Pass 1: e = leaky(asrc[src]+adst[dst]); store; segment-max into mord[dst] (ordered-uint).
__global__ void edge_logits(const int* __restrict__ srcI, const int* __restrict__ dstI,
                            int E, int Etot, const float* __restrict__ asrc,
                            const float* __restrict__ adst, float* __restrict__ ew,
                            unsigned* __restrict__ mord) {
  int e = blockIdx.x * blockDim.x + threadIdx.x;
  if (e >= Etot) return;
  int s, d;
  if (e < E) { s = srcI[e]; d = dstI[e]; } else { s = e - E; d = s; }
  const float4 av = *(const float4*)(asrc + (size_t)s * 4);
  const float4 bv = *(const float4*)(adst + (size_t)d * 4);
  float4 ev;
  ev.x = leaky02(av.x + bv.x);
  ev.y = leaky02(av.y + bv.y);
  ev.z = leaky02(av.z + bv.z);
  ev.w = leaky02(av.w + bv.w);
  *(float4*)(ew + (size_t)e * 4) = ev;
  unsigned* mp = mord + (size_t)d * 4;
  atomicMax(mp + 0, f2ord(ev.x));
  atomicMax(mp + 1, f2ord(ev.y));
  atomicMax(mp + 2, f2ord(ev.z));
  atomicMax(mp + 3, f2ord(ev.w));
}

// Pass 2: w = exp(e - max[dst]); store; segment-sum into denom[dst].
__global__ void edge_exp(const int* __restrict__ dstI, int E, int Etot,
                         const unsigned* __restrict__ mord, float* __restrict__ ew,
                         float* __restrict__ denom) {
  int e = blockIdx.x * blockDim.x + threadIdx.x;
  if (e >= Etot) return;
  int d = (e < E) ? dstI[e] : (e - E);
  float4 ev = *(const float4*)(ew + (size_t)e * 4);
  const unsigned* mp = mord + (size_t)d * 4;
  float4 wv;
  wv.x = expf(ev.x - ord2f(mp[0]));
  wv.y = expf(ev.y - ord2f(mp[1]));
  wv.z = expf(ev.z - ord2f(mp[2]));
  wv.w = expf(ev.w - ord2f(mp[3]));
  *(float4*)(ew + (size_t)e * 4) = wv;
  float* dp = denom + (size_t)d * 4;
  atomicAdd(dp + 0, wv.x);
  atomicAdd(dp + 1, wv.y);
  atomicAdd(dp + 2, wv.z);
  atomicAdd(dp + 3, wv.w);
}

// Pass 3: agg[dst] += alpha * feat[src].  Grid: (edges, HC/4 four-channel chunks).
__global__ void edge_msg(const int* __restrict__ srcI, const int* __restrict__ dstI,
                         int E, int Etot, const float* __restrict__ ew,
                         const float* __restrict__ denom, const float* __restrict__ feat,
                         float* __restrict__ agg, int HC, int C) {
  int e = blockIdx.x * blockDim.x + threadIdx.x;
  if (e >= Etot) return;
  int s, d;
  if (e < E) { s = srcI[e]; d = dstI[e]; } else { s = e - E; d = s; }
  const float4 wv = *(const float4*)(ew + (size_t)e * 4);
  const float4 dv = *(const float4*)(denom + (size_t)d * 4);
  const float alpha[4] = { wv.x / (dv.x + 1e-16f), wv.y / (dv.y + 1e-16f),
                           wv.z / (dv.z + 1e-16f), wv.w / (dv.w + 1e-16f) };
  const int f0 = blockIdx.y * 4;
  const float4 hv = *(const float4*)(feat + (size_t)s * HC + f0);
  const float vals[4] = { hv.x, hv.y, hv.z, hv.w };
  float* ap = agg + (size_t)d * HC + f0;
#pragma unroll
  for (int i = 0; i < 4; ++i) {
    int h = (f0 + i) / C;
    atomicAdd(ap + i, vals[i] * alpha[h]);
  }
}

__global__ void bias_relu(float* __restrict__ x, const float* __restrict__ b,
                          long long total, int F) {
  long long t = (long long)blockIdx.x * blockDim.x + threadIdx.x;
  if (t >= total) return;
  int f = (int)(t % F);
  float v = x[t] + b[f];
  x[t] = v > 0.f ? v : 0.f;
}

__global__ void head_mean_bias_relu(const float* __restrict__ agg, const float* __restrict__ b,
                                    float* __restrict__ out, int n, int C) {
  long long t = (long long)blockIdx.x * blockDim.x + threadIdx.x;
  if (t >= (long long)n * C) return;
  int i = (int)(t / C), c = (int)(t % C);
  const float* ap = agg + (size_t)i * GAT_HEADS * C + c;
  float s = 0.f;
#pragma unroll
  for (int h = 0; h < GAT_HEADS; ++h) s += ap[h * C];
  float v = s * (1.f / GAT_HEADS) + b[c];
  out[t] = v > 0.f ? v : 0.f;
}

// final [N,15] @ [15,2] + bias (too skinny for WMMA to matter)
__global__ void mlp_out(const float* __restrict__ hid, const float* __restrict__ w,
                        const float* __restrict__ b, float* __restrict__ out, int n) {
  int i = blockIdx.x * blockDim.x + threadIdx.x;
  if (i >= n) return;
  float a0 = b[0], a1 = b[1];
  const float* hp = hid + (size_t)i * 15;
#pragma unroll
  for (int k = 0; k < 15; ++k) {
    float h = hp[k];
    a0 += h * w[2 * k + 0];
    a1 += h * w[2 * k + 1];
  }
  out[2 * i + 0] = a0;
  out[2 * i + 1] = a1;
}

extern "C" void kernel_launch(void* const* d_in, const int* in_sizes, int n_in,
                              void* d_out, int out_size, void* d_ws, size_t ws_size,
                              hipStream_t stream) {
  const float* x   = (const float*)d_in[0];
  const int*   ei  = (const int*)d_in[1];
  /* d_in[2] edge_attr unused (edge_dim=None) */
  const float* w1  = (const float*)d_in[3];
  const float* as1 = (const float*)d_in[4];
  const float* ad1 = (const float*)d_in[5];
  const float* b1  = (const float*)d_in[6];
  const float* w2  = (const float*)d_in[7];
  const float* as2 = (const float*)d_in[8];
  const float* ad2 = (const float*)d_in[9];
  const float* b2  = (const float*)d_in[10];
  const float* fw1 = (const float*)d_in[11];
  const float* fb1 = (const float*)d_in[12];
  const float* fw2 = (const float*)d_in[13];
  const float* fb2 = (const float*)d_in[14];

  const int N = in_sizes[0] / 6;
  const int E = in_sizes[1] / 2;
  const int Etot = E + N;                 // + self loops
  const int* srcI = ei;                   // edge_index row 0
  const int* dstI = ei + E;               // edge_index row 1

  // workspace layout (float-element offsets from ws base); everything 16B-aligned
  float* ws = (float*)d_ws;
  size_t off = 0;
  const size_t o_h1    = off; off += (size_t)N * 24;
  const size_t o_asrc  = off; off += (size_t)N * 4;
  const size_t o_adst  = off; off += (size_t)N * 4;
  const size_t o_ew    = off; off += (size_t)Etot * 4;
  const size_t o_mord  = off; off += (size_t)N * 4;
  const size_t o_denom = off; off += (size_t)N * 4;
  const size_t o_agg1  = off; off += (size_t)N * 24;   // becomes h1b in place
  const size_t o_h2    = off; off += (size_t)N * 120;
  const size_t o_agg2  = off; off += (size_t)N * 120;
  const size_t o_out30 = off; off += (size_t)N * 30;
  const size_t o_hid15 = off; off += (size_t)N * 15;
  const size_t o_dump  = off; off += 64;               // sink for clamped OOB tile stores

  float*    h1    = ws + o_h1;
  float*    asrc  = ws + o_asrc;
  float*    adst  = ws + o_adst;
  float*    ew    = ws + o_ew;
  unsigned* mord  = (unsigned*)(ws + o_mord);
  float*    denom = ws + o_denom;
  float*    agg1  = ws + o_agg1;
  float*    h2    = ws + o_h2;
  float*    agg2  = ws + o_agg2;
  float*    out30 = ws + o_out30;
  float*    hid15 = ws + o_hid15;
  float*    outp  = (float*)d_out;

  auto cdiv = [](long long a, long long b) { return (int)((a + b - 1) / b); };
  const unsigned NEG_INF_ORD = 0x007FFFFFu;  // f2ord(-inf)
  const unsigned DUMP = (unsigned)o_dump;
  const int gemm_blocks = cdiv((long long)cdiv(N, 16) * 32, 256);

  // ---------------- Layer 1: GATConv(6 -> 4x6, concat) ----------------
  gemm_wmma<6, 24><<<gemm_blocks, 256, 0, stream>>>(x, w1, ws, (unsigned)o_h1, N, DUMP);
  att_coef<<<cdiv(N, 256), 256, 0, stream>>>(h1, as1, ad1, asrc, adst, N, 6);
  fill_u32<<<cdiv((long long)N * 4, 256), 256, 0, stream>>>(mord, NEG_INF_ORD, (long long)N * 4);
  fill_u32<<<cdiv((long long)N * 4, 256), 256, 0, stream>>>((unsigned*)denom, 0u, (long long)N * 4);
  fill_u32<<<cdiv((long long)N * 24, 256), 256, 0, stream>>>((unsigned*)agg1, 0u, (long long)N * 24);
  edge_logits<<<cdiv(Etot, 256), 256, 0, stream>>>(srcI, dstI, E, Etot, asrc, adst, ew, mord);
  edge_exp<<<cdiv(Etot, 256), 256, 0, stream>>>(dstI, E, Etot, mord, ew, denom);
  edge_msg<<<dim3(cdiv(Etot, 256), 6), 256, 0, stream>>>(srcI, dstI, E, Etot, ew, denom, h1, agg1, 24, 6);
  bias_relu<<<cdiv((long long)N * 24, 256), 256, 0, stream>>>(agg1, b1, (long long)N * 24, 24);

  // ---------------- Layer 2: GATConv(24 -> 4x30, head-mean) ----------------
  gemm_wmma<24, 120><<<gemm_blocks, 256, 0, stream>>>(agg1, w2, ws, (unsigned)o_h2, N, DUMP);
  att_coef<<<cdiv(N, 256), 256, 0, stream>>>(h2, as2, ad2, asrc, adst, N, 30);
  fill_u32<<<cdiv((long long)N * 4, 256), 256, 0, stream>>>(mord, NEG_INF_ORD, (long long)N * 4);
  fill_u32<<<cdiv((long long)N * 4, 256), 256, 0, stream>>>((unsigned*)denom, 0u, (long long)N * 4);
  fill_u32<<<cdiv((long long)N * 120, 256), 256, 0, stream>>>((unsigned*)agg2, 0u, (long long)N * 120);
  edge_logits<<<cdiv(Etot, 256), 256, 0, stream>>>(srcI, dstI, E, Etot, asrc, adst, ew, mord);
  edge_exp<<<cdiv(Etot, 256), 256, 0, stream>>>(dstI, E, Etot, mord, ew, denom);
  edge_msg<<<dim3(cdiv(Etot, 256), 30), 256, 0, stream>>>(srcI, dstI, E, Etot, ew, denom, h2, agg2, 120, 30);
  head_mean_bias_relu<<<cdiv((long long)N * 30, 256), 256, 0, stream>>>(agg2, b2, out30, N, 30);

  // ---------------- MLP head ----------------
  gemm_wmma<30, 15><<<gemm_blocks, 256, 0, stream>>>(out30, fw1, ws, (unsigned)o_hid15, N, DUMP);
  bias_relu<<<cdiv((long long)N * 15, 256), 256, 0, stream>>>(hid15, fb1, (long long)N * 15, 15);
  mlp_out<<<cdiv(N, 256), 256, 0, stream>>>(hid15, fw2, fb2, outp, N);

  (void)n_in; (void)out_size; (void)ws_size;
}
